// DnnSepParser_90924457656670
// MI455X (gfx1250) — compile-verified, matrix operand order checked by reading, not run on hardware
//
#include <hip/hip_runtime.h>
#include <hip/hip_bf16.h>

typedef unsigned short u16;
typedef unsigned int   u32;
typedef __attribute__((ext_vector_type(16))) __bf16 v16bf;
typedef __attribute__((ext_vector_type(8)))  float  v8f;

#define NTOK 1024
#define HDIM 128
#define G4H  512
#define WD   100
#define TD   28
#define WHH_STRIDE 66   // dwords per gate row in LDS: 64 data + 2 pad (bank-conflict free)

__device__ __forceinline__ u16 f2bf(float f) {
  u32 u = __float_as_uint(f);
  u32 r = u + 0x7FFFu + ((u >> 16) & 1u);   // round-to-nearest-even
  return (u16)(r >> 16);
}
__device__ __forceinline__ float bf2f(u16 h) {
  return __uint_as_float(((u32)h) << 16);
}
__device__ __forceinline__ float sigmoidf_(float x) {
  return 1.f / (1.f + __expf(-x));
}
__device__ __forceinline__ float tanhf_(float x) {
  float e = __expf(2.f * x);
  return 1.f - 2.f / (e + 1.f);
}

// ---------------------------------------------------------------------------
// Embedding gather + concat -> x0 (1024 x 128) bf16
// ---------------------------------------------------------------------------
__global__ void embed_kernel(const int* __restrict__ widx, const int* __restrict__ tidx,
                             const float* __restrict__ wemb, const float* __restrict__ temb,
                             u16* __restrict__ x0b) {
  int i = blockIdx.x;
  int d = threadIdx.x;  // 0..127
  float val;
  if (d < WD) val = wemb[(long)widx[i] * WD + d];
  else        val = temb[(long)tidx[i] * TD + (d - WD)];
  x0b[i * (WD + TD) + d] = f2bf(val);
}

__global__ void f32_to_bf16_kernel(const float* __restrict__ src, u16* __restrict__ dst, int n) {
  int i = blockIdx.x * blockDim.x + threadIdx.x;
  if (i < n) dst[i] = f2bf(src[i]);
}

// ---------------------------------------------------------------------------
// C[M x ncols] = A[M x K](bf16) * W[nrows x K](bf16)^T + bias1 + bias2
// One wave (32 threads) per 16x16 output tile; v_wmma_f32_16x16x32_bf16.
// Fragment layout per CDNA5 ISA 7.12.2:
//   lanes 0-15 : row (A) / col (B) = lane,    K = {kb..kb+7, kb+16..kb+23}
//   lanes 16-31: row/col = lane-16,           K = {kb+8..kb+15, kb+24..kb+31}
//   D vgpr r   : lanes 0-15 -> M=m0+r, lanes 16-31 -> M=m0+r+8, N = lane&15
// ---------------------------------------------------------------------------
union Frag16 { u32 w[8]; v16bf v; };

__global__ void gemm_bf16_wmma(const u16* __restrict__ A, int lda,
                               const u16* __restrict__ W, int ldw, int nrows,
                               const float* __restrict__ bias1,
                               const float* __restrict__ bias2,
                               float* __restrict__ C, int ldc, int ncols, int K) {
  int lane = threadIdx.x;
  int m0 = blockIdx.x * 16, n0 = blockIdx.y * 16;
  int hl  = lane >> 4;        // 0 or 1 (K-half select)
  int sub = lane & 15;
  int offs = hl * 8;

  const u16* Arow = A + (long)(m0 + sub) * lda + offs;
  int nIdx = n0 + sub;
  int nClamp = (nIdx < nrows) ? nIdx : (nrows - 1);
  const u16* Wrow = W + (long)nClamp * ldw + offs;

  v8f acc = {};
  for (int kb = 0; kb < K; kb += 32) {
    Frag16 a, b;
    *(uint4*)&a.w[0] = *(const uint4*)(Arow + kb);
    *(uint4*)&a.w[4] = *(const uint4*)(Arow + kb + 16);
    *(uint4*)&b.w[0] = *(const uint4*)(Wrow + kb);
    *(uint4*)&b.w[4] = *(const uint4*)(Wrow + kb + 16);
    acc = __builtin_amdgcn_wmma_f32_16x16x32_bf16(false, a.v, false, b.v,
                                                  (short)0, acc, false, false);
  }

  if (nIdx < ncols) {
    float badd = 0.f;
    if (bias1) badd += bias1[nIdx];
    if (bias2) badd += bias2[nIdx];
#pragma unroll
    for (int r = 0; r < 8; ++r) {
      int m = m0 + r + hl * 8;
      C[(long)m * ldc + nIdx] = acc[r] + badd;
    }
  }
}

// ---------------------------------------------------------------------------
// Sequential LSTM scan, one 512-thread block per direction (blockIdx.x: 0=fwd,
// 1=bwd). Whh (512x128) resident in LDS as packed bf16 pairs, padded stride.
// g[tid] = xg[t][tid] + dot128(h, Whh[tid]); gate math by threads 0..127.
// Writes h_t (bf16) into out[t][colbase + 0..127], out_ld = 256 (concat).
// ---------------------------------------------------------------------------
__global__ void lstm_scan_kernel(const float* __restrict__ xg_f,
                                 const float* __restrict__ xg_b,
                                 const float* __restrict__ whh_f,
                                 const float* __restrict__ whh_b,
                                 u16* __restrict__ out_b, int out_ld) {
  extern __shared__ char smem[];
  u32*   whh_lds = (u32*)smem;                                   // 512 * 66 dwords
  float* hbuf    = (float*)(smem + (size_t)G4H * WHH_STRIDE * 4); // 128
  float* cbuf    = hbuf + HDIM;                                   // 128
  float* gbuf    = cbuf + HDIM;                                   // 512

  int dir = blockIdx.x;
  const float* xg  = dir ? xg_b  : xg_f;
  const float* whh = dir ? whh_b : whh_f;
  int colbase = dir ? HDIM : 0;
  int tid = threadIdx.x;  // 0..511

  // Stage Whh into LDS as bf16x2
  for (int idx = tid; idx < G4H * (HDIM / 2); idx += blockDim.x) {
    int row = idx / (HDIM / 2);
    int k2  = idx % (HDIM / 2);
    u32 lo = (u32)f2bf(whh[row * HDIM + 2 * k2]);
    u32 hi = (u32)f2bf(whh[row * HDIM + 2 * k2 + 1]);
    whh_lds[row * WHH_STRIDE + k2] = lo | (hi << 16);
  }
  if (tid < HDIM) { hbuf[tid] = 0.f; cbuf[tid] = 0.f; }
  __syncthreads();

  const u32* wrow = whh_lds + tid * WHH_STRIDE;
  for (int s = 0; s < NTOK; ++s) {
    int t = dir ? (NTOK - 1 - s) : s;
    float g = xg[(long)t * G4H + tid];
#pragma unroll 8
    for (int k2 = 0; k2 < HDIM / 2; ++k2) {
      u32 w = wrow[k2];
      g += bf2f((u16)(w & 0xFFFFu)) * hbuf[2 * k2];
      g += bf2f((u16)(w >> 16))     * hbuf[2 * k2 + 1];
    }
    gbuf[tid] = g;
    __syncthreads();
    if (tid < HDIM) {
      float gi = gbuf[tid];
      float gf = gbuf[HDIM + tid];
      float gg = gbuf[2 * HDIM + tid];
      float go = gbuf[3 * HDIM + tid];
      float c2 = sigmoidf_(gf) * cbuf[tid] + sigmoidf_(gi) * tanhf_(gg);
      float h2 = sigmoidf_(go) * tanhf_(c2);
      cbuf[tid] = c2;
      hbuf[tid] = h2;
      out_b[(long)t * out_ld + colbase + tid] = f2bf(h2);
    }
    __syncthreads();
  }
}

// ---------------------------------------------------------------------------
// Fused: scores[i][j] = sum_k tanh(u[i][k]+v[j][k])*w2[k] + b2, then
// log_softmax over j. One block per row i (256 threads, 4 j's each).
// ---------------------------------------------------------------------------
__global__ void score_logsoftmax_kernel(const float* __restrict__ u,
                                        const float* __restrict__ v,
                                        const float* __restrict__ w2,
                                        const float* __restrict__ b2,
                                        float* __restrict__ out) {
  __shared__ float us[WD];
  __shared__ float ws[WD];
  __shared__ float srow[NTOK];
  __shared__ float red[256];
  int i = blockIdx.x;
  int tid = threadIdx.x;
  if (tid < WD) { us[tid] = u[i * WD + tid]; ws[tid] = w2[tid]; }
  __syncthreads();
  float bias = b2[0];
#pragma unroll
  for (int jj = 0; jj < 4; ++jj) {
    int j = jj * 256 + tid;
    const float* vj = v + (long)j * WD;
    float acc = 0.f;
#pragma unroll 4
    for (int k = 0; k < WD; ++k)
      acc += tanhf_(us[k] + vj[k]) * ws[k];
    srow[j] = acc + bias;
  }
  __syncthreads();
  float m = -1e30f;
#pragma unroll
  for (int jj = 0; jj < 4; ++jj) m = fmaxf(m, srow[jj * 256 + tid]);
  red[tid] = m;
  __syncthreads();
  for (int off = 128; off > 0; off >>= 1) {
    if (tid < off) red[tid] = fmaxf(red[tid], red[tid + off]);
    __syncthreads();
  }
  float rmax = red[0];
  __syncthreads();
  float se = 0.f;
#pragma unroll
  for (int jj = 0; jj < 4; ++jj) se += __expf(srow[jj * 256 + tid] - rmax);
  red[tid] = se;
  __syncthreads();
  for (int off = 128; off > 0; off >>= 1) {
    if (tid < off) red[tid] += red[tid + off];
    __syncthreads();
  }
  float lse = __logf(red[0]) + rmax;
#pragma unroll
  for (int jj = 0; jj < 4; ++jj) {
    int j = jj * 256 + tid;
    out[(long)i * NTOK + j] = srow[j] - lse;
  }
}

// ---------------------------------------------------------------------------
extern "C" void kernel_launch(void* const* d_in, const int* in_sizes, int n_in,
                              void* d_out, int out_size, void* d_ws, size_t ws_size,
                              hipStream_t stream) {
  const int*   widx = (const int*)d_in[0];
  const int*   tidx = (const int*)d_in[1];
  const float* wemb = (const float*)d_in[2];
  const float* temb = (const float*)d_in[3];
  // lstm_params flattened: (L0 fwd)(L0 bwd)(L1 fwd)(L1 bwd), each Wih,Whh,bih,bhh
  const float* Wih[4] = {(const float*)d_in[4],  (const float*)d_in[8],
                         (const float*)d_in[12], (const float*)d_in[16]};
  const float* Whh[4] = {(const float*)d_in[5],  (const float*)d_in[9],
                         (const float*)d_in[13], (const float*)d_in[17]};
  const float* bih[4] = {(const float*)d_in[6],  (const float*)d_in[10],
                         (const float*)d_in[14], (const float*)d_in[18]};
  const float* bhh[4] = {(const float*)d_in[7],  (const float*)d_in[11],
                         (const float*)d_in[15], (const float*)d_in[19]};
  const float* W1 = (const float*)d_in[20];
  const float* b1 = (const float*)d_in[21];
  const float* W2 = (const float*)d_in[22];
  const float* b2 = (const float*)d_in[23];
  float* out = (float*)d_out;

  // Workspace carve-out (~7 MB)
  char* ws = (char*)d_ws;
  size_t off = 0;
  auto alloc = [&](size_t bytes) -> char* {
    char* p = ws + off;
    off += (bytes + 255) & ~(size_t)255;
    return p;
  };
  u16* x0b = (u16*)alloc((size_t)NTOK * 128 * 2);
  u16* wihb[4];
  wihb[0] = (u16*)alloc((size_t)512 * 128 * 2);
  wihb[1] = (u16*)alloc((size_t)512 * 128 * 2);
  wihb[2] = (u16*)alloc((size_t)512 * 256 * 2);
  wihb[3] = (u16*)alloc((size_t)512 * 256 * 2);
  u16*   w1b = (u16*)alloc((size_t)100 * 512 * 2);
  float* xgf = (float*)alloc((size_t)NTOK * 512 * 4);
  float* xgb = (float*)alloc((size_t)NTOK * 512 * 4);
  u16*   x1b = (u16*)alloc((size_t)NTOK * 256 * 2);
  u16*   x2b = (u16*)alloc((size_t)NTOK * 256 * 2);
  float* ub  = (float*)alloc((size_t)NTOK * 100 * 4);
  float* vb  = (float*)alloc((size_t)NTOK * 100 * 4);
  (void)ws_size; (void)in_sizes; (void)n_in; (void)out_size;

  size_t smem_scan = (size_t)G4H * WHH_STRIDE * 4 + (HDIM + HDIM + G4H) * 4;
  hipFuncSetAttribute((const void*)lstm_scan_kernel,
                      hipFuncAttributeMaxDynamicSharedMemorySize, (int)smem_scan);

  embed_kernel<<<NTOK, 128, 0, stream>>>(widx, tidx, wemb, temb, x0b);
  f32_to_bf16_kernel<<<(512 * 128 + 255) / 256, 256, 0, stream>>>(Wih[0], wihb[0], 512 * 128);
  f32_to_bf16_kernel<<<(512 * 128 + 255) / 256, 256, 0, stream>>>(Wih[1], wihb[1], 512 * 128);
  f32_to_bf16_kernel<<<(512 * 256 + 255) / 256, 256, 0, stream>>>(Wih[2], wihb[2], 512 * 256);
  f32_to_bf16_kernel<<<(512 * 256 + 255) / 256, 256, 0, stream>>>(Wih[3], wihb[3], 512 * 256);
  f32_to_bf16_kernel<<<(100 * 512 + 255) / 256, 256, 0, stream>>>(W1, w1b, 100 * 512);

  // Layer 0: xg = x0 * Wih^T + (bih+bhh), then bidirectional scan -> x1
  gemm_bf16_wmma<<<dim3(64, 32), 32, 0, stream>>>(x0b, 128, wihb[0], 128, 512,
                                                  bih[0], bhh[0], xgf, 512, 512, 128);
  gemm_bf16_wmma<<<dim3(64, 32), 32, 0, stream>>>(x0b, 128, wihb[1], 128, 512,
                                                  bih[1], bhh[1], xgb, 512, 512, 128);
  lstm_scan_kernel<<<2, 512, smem_scan, stream>>>(xgf, xgb, Whh[0], Whh[1], x1b, 256);

  // Layer 1
  gemm_bf16_wmma<<<dim3(64, 32), 32, 0, stream>>>(x1b, 256, wihb[2], 256, 512,
                                                  bih[2], bhh[2], xgf, 512, 512, 256);
  gemm_bf16_wmma<<<dim3(64, 32), 32, 0, stream>>>(x1b, 256, wihb[3], 256, 512,
                                                  bih[3], bhh[3], xgb, 512, 512, 256);
  lstm_scan_kernel<<<2, 512, smem_scan, stream>>>(xgf, xgb, Whh[2], Whh[3], x2b, 256);

  // u = x2*A^T + b1 ; v = x2*B^T   (A = W1[:, :256], B = W1[:, 256:])
  gemm_bf16_wmma<<<dim3(64, 7), 32, 0, stream>>>(x2b, 256, w1b, 512, 100,
                                                 b1, nullptr, ub, 100, 100, 256);
  gemm_bf16_wmma<<<dim3(64, 7), 32, 0, stream>>>(x2b, 256, w1b + 256, 512, 100,
                                                 nullptr, nullptr, vb, 100, 100, 256);

  // Fused biaffine scores + log_softmax(axis=1)
  score_logsoftmax_kernel<<<NTOK, 256, 0, stream>>>(ub, vb, W2, b2, out);
}
